// CCRNN_37306085933744
// MI455X (gfx1250) — matrix operand order repaired and checked by reading, not compiled, not run on hardware
//
#include <hip/hip_runtime.h>
#include <math.h>

// ---------------- model constants ----------------
#define NN     512      // nodes
#define ND     64       // node-embedding dim
#define HD     64       // hidden
#define BB     16       // batch
#define TT     12       // encoder steps
#define NPRED  12       // decoder steps
#define D0     65       // layer-0 feature width (1 + HD)
#define D0PAD  80       // padded layer-0 width (multiple of 16)
#define FBLD   128      // max plane width; fb node stride = 4*dpad <= 4*FBLD
#define TPL    3        // transposed planes kept (x0,x1,x2 serve as B operands)

typedef __attribute__((ext_vector_type(16))) __bf16 v16bf;
typedef __attribute__((ext_vector_type(8)))  __bf16 v8bf;
typedef __attribute__((ext_vector_type(8)))  float  v8f;

__device__ __forceinline__ float sigmoidf_(float x){ return 1.f/(1.f+expf(-x)); }

__device__ __forceinline__ v16bf cat16(v8bf a, v8bf b) {
  return __builtin_shufflevector(a, b, 0,1,2,3,4,5,6,7,8,9,10,11,12,13,14,15);
}

// fbT layout: fbT[((b*TPL + k)*FBLD + jc)*NN + node]   (node contiguous)
__device__ __forceinline__ size_t tidx(int b, int k, int jc, int node) {
  return (((size_t)b * TPL + k) * FBLD + jc) * NN + node;
}

// =====================================================================
// nodevec stage update: nv1' = nv1 @ w + b ; nv2'[d,n] = sum_k nv2[k,n]*w[k,d] + b[d]
// =====================================================================
__global__ void nv_stage_kernel(const float* __restrict__ nv1_in,
                                const float* __restrict__ nv2_in,
                                const float* __restrict__ w,
                                const float* __restrict__ bvec,
                                float* __restrict__ nv1_out,
                                float* __restrict__ nv2_out) {
  int idx = blockIdx.x * blockDim.x + threadIdx.x;
  if (idx < NN * ND) {
    int m = idx / ND, d = idx % ND;
    float acc = bvec[d];
    for (int k = 0; k < ND; ++k) acc += nv1_in[m * ND + k] * w[k * ND + d];
    nv1_out[idx] = acc;
  } else if (idx < 2 * NN * ND) {
    idx -= NN * ND;
    int d = idx / NN, n = idx % NN;
    float acc = bvec[d];
    for (int k = 0; k < ND; ++k) acc += nv2_in[k * NN + n] * w[k * ND + d];
    nv2_out[d * NN + n] = acc;
  }
}

// =====================================================================
// support row: g[m,:] = softmax(leaky_relu(nv1[m,:] @ nv2)) -> bf16
// =====================================================================
__global__ void support_kernel(const float* __restrict__ nv1,
                               const float* __restrict__ nv2,
                               __bf16* __restrict__ Sout) {
  __shared__ float red[256];
  __shared__ float zrow[NN];
  int m = blockIdx.x;
  const float* a = nv1 + (size_t)m * ND;
  float lmax = -3.0e38f;
  for (int n = threadIdx.x; n < NN; n += 256) {
    float acc = 0.f;
    #pragma unroll 8
    for (int k = 0; k < ND; ++k) acc += a[k] * nv2[k * NN + n];
    acc = acc > 0.f ? acc : 0.01f * acc;     // leaky_relu
    zrow[n] = acc;
    lmax = fmaxf(lmax, acc);
  }
  red[threadIdx.x] = lmax; __syncthreads();
  for (int s = 128; s > 0; s >>= 1) {
    if (threadIdx.x < s) red[threadIdx.x] = fmaxf(red[threadIdx.x], red[threadIdx.x + s]);
    __syncthreads();
  }
  float rmax = red[0]; __syncthreads();
  float lsum = 0.f;
  for (int n = threadIdx.x; n < NN; n += 256) {
    float e = expf(zrow[n] - rmax);
    zrow[n] = e; lsum += e;
  }
  red[threadIdx.x] = lsum; __syncthreads();
  for (int s = 128; s > 0; s >>= 1) {
    if (threadIdx.x < s) red[threadIdx.x] += red[threadIdx.x + s];
    __syncthreads();
  }
  float inv = 1.f / red[0];
  for (int n = threadIdx.x; n < NN; n += 256)
    Sout[(size_t)m * NN + n] = (__bf16)(zrow[n] * inv);
}

// =====================================================================
// Chebyshev propagation, barrier-free, 2xM register-blocked:
//   plane[kout] = alpha * S @ plane[kx] + beta * plane[kz]
// Each wave computes two 16x16 M-tiles sharing one B operand
// (3 b128-load pairs per 2 WMMAs). A = S rows; B/Z = transposed planes.
// grid = (NN/128, dpad/16, B), block = 128 (4 waves x 2 M-tiles = 128 rows)
// =====================================================================
__global__ void cheb_gemm_kernel(const __bf16* __restrict__ S,
                                 __bf16* __restrict__ fb,    // plane layout (write)
                                 __bf16* __restrict__ fbT,   // transposed (read kx,kz / write kout)
                                 int kx, int kz, int kout,
                                 float alpha, float beta, int dpad) {
  const int lane = threadIdx.x & 31;
  const int wave = threadIdx.x >> 5;
  const int b    = blockIdx.z;
  const int n0   = blockIdx.y * 16;
  const int m0   = blockIdx.x * 128 + wave * 32;   // two tiles: m0, m0+16
  const int kgrp = lane >> 4;
  const int lid  = lane & 15;
  const int col  = n0 + lid;

  const __bf16* Bcol = fbT + tidx(b, kx, col, 0) + kgrp * 8;  // B: K runs along nodes
  const __bf16* Sx0  = S + (size_t)(m0 + lid) * NN + kgrp * 8;
  const __bf16* Sx1  = S + (size_t)(m0 + 16 + lid) * NN + kgrp * 8;

  v8f acc0 = {0.f,0.f,0.f,0.f,0.f,0.f,0.f,0.f};
  v8f acc1 = {0.f,0.f,0.f,0.f,0.f,0.f,0.f,0.f};

  #pragma unroll 2
  for (int k0 = 0; k0 < NN; k0 += 32) {
    if (k0 + 32 < NN) {
      __builtin_prefetch(Sx0 + k0 + 32, 0, 1);     // global_prefetch_b8
      __builtin_prefetch(Sx1 + k0 + 32, 0, 1);
    }
    v16bf bv  = cat16(*(const v8bf*)(Bcol + k0),
                      *(const v8bf*)(Bcol + k0 + 16));
    v16bf av0 = cat16(*(const v8bf*)(Sx0 + k0),
                      *(const v8bf*)(Sx0 + k0 + 16));
    v16bf av1 = cat16(*(const v8bf*)(Sx1 + k0),
                      *(const v8bf*)(Sx1 + k0 + 16));
    acc0 = __builtin_amdgcn_wmma_f32_16x16x32_bf16(false, av0, false, bv,
                                                   (short)0, acc0, false, false);
    acc1 = __builtin_amdgcn_wmma_f32_16x16x32_bf16(false, av1, false, bv,
                                                   (short)0, acc1, false, false);
  }

  const size_t ns = (size_t)4 * dpad;
  #pragma unroll
  for (int t = 0; t < 2; ++t) {
    const v8f& acc = t ? acc1 : acc0;
    int mb = m0 + t * 16 + 8 * kgrp;               // C/D layout base (M = mb + r)
    v8f vals;
    if (kz >= 0) {
      v8bf zv = *(const v8bf*)(fbT + tidx(b, kz, col, mb));
      #pragma unroll
      for (int r = 0; r < 8; ++r) vals[r] = alpha * acc[r] + beta * (float)zv[r];
    } else {
      #pragma unroll
      for (int r = 0; r < 8; ++r) vals[r] = alpha * acc[r];
    }
    // plane-layout write (feat A operand), pads self-propagate zeros
    #pragma unroll
    for (int r = 0; r < 8; ++r)
      fb[((size_t)b * NN + mb + r) * ns + (size_t)kout * dpad + col] = (__bf16)vals[r];
    // transposed write: one packed 16B store
    if (kout < TPL) {
      v8bf tv;
      #pragma unroll
      for (int r = 0; r < 8; ++r) tv[r] = (__bf16)vals[r];
      *(v8bf*)(fbT + tidx(b, kout, col, mb)) = tv;
    }
  }
}

// =====================================================================
// Feature GEMM, 2xM register-blocked (shared Wt B operand):
//   out[b,m,o] = sum_{r'} fb[b,m,r'] * Wt[o,r'] + bias[o],  r' = k*dpad + jc
// Writes f32 attention-stack slot g, bf16 plane-0 of fbnext (+ transposed).
// grid = (NN/128, OD/16, B), block = 128
// =====================================================================
__global__ void feat_gemm_kernel(const __bf16* __restrict__ fb,
                                 const __bf16* __restrict__ Wt,
                                 const float* __restrict__ bias,
                                 int dpad, int OD,
                                 __bf16* __restrict__ fbnext, int dpad_next,
                                 __bf16* __restrict__ fbTnext,
                                 float* __restrict__ Sstack, int g) {
  const int lane = threadIdx.x & 31;
  const int wave = threadIdx.x >> 5;
  const int b    = blockIdx.z;
  const int o0   = blockIdx.y * 16;
  const int m0   = blockIdx.x * 128 + wave * 32;
  const int kgrp = lane >> 4;
  const int lid  = lane & 15;
  const int ocol = o0 + lid;
  const int K4   = 4 * dpad;                       // multiple of 32
  const __bf16* Arow0 = fb + ((size_t)b * NN + m0 + lid) * K4 + kgrp * 8;
  const __bf16* Arow1 = fb + ((size_t)b * NN + m0 + 16 + lid) * K4 + kgrp * 8;
  const __bf16* Wrow  = Wt + (size_t)ocol * K4 + kgrp * 8;

  v8f acc0 = {0.f,0.f,0.f,0.f,0.f,0.f,0.f,0.f};
  v8f acc1 = {0.f,0.f,0.f,0.f,0.f,0.f,0.f,0.f};

  #pragma unroll 2
  for (int r0 = 0; r0 < K4; r0 += 32) {
    v16bf bv  = cat16(*(const v8bf*)(Wrow + r0),  *(const v8bf*)(Wrow + r0 + 16));
    v16bf av0 = cat16(*(const v8bf*)(Arow0 + r0), *(const v8bf*)(Arow0 + r0 + 16));
    v16bf av1 = cat16(*(const v8bf*)(Arow1 + r0), *(const v8bf*)(Arow1 + r0 + 16));
    acc0 = __builtin_amdgcn_wmma_f32_16x16x32_bf16(false, av0, false, bv,
                                                   (short)0, acc0, false, false);
    acc1 = __builtin_amdgcn_wmma_f32_16x16x32_bf16(false, av1, false, bv,
                                                   (short)0, acc1, false, false);
  }

  float bo = bias[ocol];
  #pragma unroll
  for (int t = 0; t < 2; ++t) {
    const v8f& acc = t ? acc1 : acc0;
    int mb = m0 + t * 16 + 8 * kgrp;
    v8bf tv;
    #pragma unroll
    for (int r = 0; r < 8; ++r) {
      int m = mb + r;
      float v = acc[r] + bo;
      Sstack[(((size_t)b * 3 + g) * NN + m) * (size_t)OD + ocol] = v;
      fbnext[(((size_t)b * NN + m) * 4) * (size_t)dpad_next + ocol] = (__bf16)v;
      tv[r] = (__bf16)v;
    }
    *(v8bf*)(fbTnext + tidx(b, 0, ocol, mb)) = tv;
  }
}

// =====================================================================
// weight pack: Wt[o][k*dpad+jc] = (jc<d) ? W[(jc*4+k)*OD + o] : 0   (bf16)
// =====================================================================
__global__ void wt_pack_kernel(const float* __restrict__ W, __bf16* __restrict__ Wt,
                               int d, int dpad, int OD) {
  int idx = blockIdx.x * blockDim.x + threadIdx.x;
  int K4 = 4 * dpad;
  if (idx >= OD * K4) return;
  int o = idx / K4, r = idx % K4;
  int k = r / dpad, jc = r % dpad;
  float v = (jc < d) ? W[((size_t)jc * 4 + k) * OD + o] : 0.f;
  Wt[idx] = (__bf16)v;
}

// =====================================================================
// attention logits: logits[b,g] = sum(Sstack[b,g,:,:] * attW) + attb
// =====================================================================
__global__ void att_logits_kernel(const float* __restrict__ Sstack,
                                  const float* __restrict__ attW,
                                  const float* __restrict__ attb,
                                  float* __restrict__ logits, int OD) {
  __shared__ float red[256];
  int bg = blockIdx.x;
  const float* base = Sstack + (size_t)bg * NN * OD;
  size_t total = (size_t)NN * OD;
  float acc = 0.f;
  for (size_t i = threadIdx.x; i < total; i += 256) acc += base[i] * attW[i];
  red[threadIdx.x] = acc; __syncthreads();
  for (int s = 128; s > 0; s >>= 1) {
    if (threadIdx.x < s) red[threadIdx.x] += red[threadIdx.x + s];
    __syncthreads();
  }
  if (threadIdx.x == 0) logits[bg] = red[0] + attb[0];
}

// softmax over g (3) and weighted sum -> compact f32 (B,NN,OD)
__global__ void att_combine_kernel(const float* __restrict__ Sstack,
                                   const float* __restrict__ logits,
                                   float* __restrict__ out, int OD) {
  size_t idx = (size_t)blockIdx.x * blockDim.x + threadIdx.x;
  size_t total = (size_t)BB * NN * OD;
  if (idx >= total) return;
  int b = (int)(idx / ((size_t)NN * OD));
  size_t rem = idx % ((size_t)NN * OD);
  float l0 = logits[b * 3 + 0], l1 = logits[b * 3 + 1], l2 = logits[b * 3 + 2];
  float mx = fmaxf(l0, fmaxf(l1, l2));
  float e0 = expf(l0 - mx), e1 = expf(l1 - mx), e2 = expf(l2 - mx);
  float inv = 1.f / (e0 + e1 + e2);
  float s0 = Sstack[((size_t)(b * 3 + 0)) * NN * OD + rem];
  float s1 = Sstack[((size_t)(b * 3 + 1)) * NN * OD + rem];
  float s2 = Sstack[((size_t)(b * 3 + 2)) * NN * OD + rem];
  out[idx] = (s0 * e0 + s1 * e1 + s2 * e2) * inv;
}

// =====================================================================
// pointwise GRU kernels (write bf16 plane-0 of fb + fbT, dpad = D0PAD)
// =====================================================================
__global__ void concat_in_kernel(const float* __restrict__ x, size_t xbs,
                                 const float* __restrict__ state,
                                 __bf16* __restrict__ fb,
                                 __bf16* __restrict__ fbT) {
  int bn = blockIdx.x * blockDim.x + threadIdx.x;
  if (bn >= BB * NN) return;
  int b = bn / NN, n = bn % NN;
  __bf16* row = fb + (size_t)bn * 4 * D0PAD;       // plane 0
  float xv = x[(size_t)b * xbs + n];
  row[0] = (__bf16)xv;
  fbT[tidx(b, 0, 0, n)] = (__bf16)xv;
  const float* st = state + (size_t)bn * HD;
  #pragma unroll 8
  for (int h = 0; h < HD; ++h) {
    __bf16 v = (__bf16)st[h];
    row[1 + h] = v;
    fbT[tidx(b, 0, 1 + h, n)] = v;
  }
  #pragma unroll
  for (int jc = D0; jc < D0PAD; ++jc) {
    row[jc] = (__bf16)0.f;
    fbT[tidx(b, 0, jc, n)] = (__bf16)0.f;
  }
}

// after ru evo: r,u = sigmoid; plane0 = [x, r*state]; ubuf = u
__global__ void gru_ru_kernel(const float* __restrict__ evo_ru,
                              const float* __restrict__ state,
                              const float* __restrict__ x, size_t xbs,
                              __bf16* __restrict__ fb,
                              __bf16* __restrict__ fbT,
                              float* __restrict__ ubuf) {
  size_t idx = (size_t)blockIdx.x * blockDim.x + threadIdx.x;
  if (idx >= (size_t)BB * NN * HD) return;
  size_t bn = idx / HD; int h = (int)(idx % HD);
  int b = (int)(bn / NN), n = (int)(bn % NN);
  float r = sigmoidf_(evo_ru[bn * 128 + h]);
  float u = sigmoidf_(evo_ru[bn * 128 + HD + h]);
  ubuf[idx] = u;
  __bf16 rs = (__bf16)(r * state[idx]);
  fb[bn * 4 * D0PAD + 1 + h] = rs;
  fbT[tidx(b, 0, 1 + h, n)] = rs;
  if (h == 0) {
    __bf16 xv = (__bf16)x[(size_t)b * xbs + n];
    fb[bn * 4 * D0PAD] = xv;
    fbT[tidx(b, 0, 0, n)] = xv;
    #pragma unroll
    for (int jc = D0; jc < D0PAD; ++jc) {
      fb[bn * 4 * D0PAD + jc] = (__bf16)0.f;
      fbT[tidx(b, 0, jc, n)] = (__bf16)0.f;
    }
  }
}

// state = u*state + (1-u)*tanh(c)
__global__ void gru_update_kernel(const float* __restrict__ evo_c,
                                  const float* __restrict__ ubuf,
                                  float* __restrict__ state) {
  size_t idx = (size_t)blockIdx.x * blockDim.x + threadIdx.x;
  if (idx >= (size_t)BB * NN * HD) return;
  float c = tanhf(evo_c[idx]);
  float u = ubuf[idx];
  state[idx] = u * state[idx] + (1.f - u) * c;
}

// decoder projection: out = h @ outW + outb ; also next decoder input
__global__ void dec_out_kernel(const float* __restrict__ state,
                               const float* __restrict__ outW,
                               const float* __restrict__ outb,
                               float* __restrict__ dout,
                               float* __restrict__ dec_in, int t) {
  int bn = blockIdx.x * blockDim.x + threadIdx.x;
  if (bn >= BB * NN) return;
  int b = bn / NN, n = bn % NN;
  float acc = outb[0];
  const float* h = state + (size_t)bn * HD;
  #pragma unroll 8
  for (int k = 0; k < HD; ++k) acc += h[k] * outW[k];
  dout[((size_t)b * NPRED + t) * NN + n] = acc;
  dec_in[bn] = acc;
}

// =====================================================================
// host orchestration
// =====================================================================
struct EvoP {
  const float* W[3]; const float* bvec[3];
  const float* attW; const float* attb;
  __bf16* Wt[3]; int d[3]; int dpad[3];
};

static void run_evo(hipStream_t st, const __bf16* const* Sbf,
                    __bf16* fb0, __bf16* fb1, __bf16* fbT0, __bf16* fbT1,
                    int OD, const EvoP& p,
                    float* sstack, float* logits, float* evo_out) {
  __bf16* fbs[2]  = {fb0, fb1};
  __bf16* fbts[2] = {fbT0, fbT1};
  for (int i = 0; i < 3; ++i) {
    int dpad = p.dpad[i];
    __bf16* fb   = fbs[i & 1];
    __bf16* fbn  = fbs[(i + 1) & 1];
    __bf16* fbt  = fbts[i & 1];
    __bf16* fbtn = fbts[(i + 1) & 1];
    dim3 gc(NN / 128, dpad / 16, BB);
    cheb_gemm_kernel<<<gc, 128, 0, st>>>(Sbf[i], fb, fbt, 0, -1, 1, 1.f,  0.f, dpad);
    cheb_gemm_kernel<<<gc, 128, 0, st>>>(Sbf[i], fb, fbt, 1,  0, 2, 2.f, -1.f, dpad);
    cheb_gemm_kernel<<<gc, 128, 0, st>>>(Sbf[i], fb, fbt, 2,  1, 3, 2.f, -1.f, dpad);
    dim3 gf(NN / 128, OD / 16, BB);
    feat_gemm_kernel<<<gf, 128, 0, st>>>(fb, p.Wt[i], p.bvec[i], dpad, OD,
                                         fbn, OD, fbtn, sstack, i);
  }
  att_logits_kernel<<<BB * 3, 256, 0, st>>>(sstack, p.attW, p.attb, logits, OD);
  size_t total = (size_t)BB * NN * OD;
  att_combine_kernel<<<(unsigned)((total + 255) / 256), 256, 0, st>>>(sstack, logits, evo_out, OD);
}

static void run_dcgru(hipStream_t st, const __bf16* const* Sbf,
                      const float* x, size_t xbs, float* state,
                      const EvoP& pru, const EvoP& pcand,
                      __bf16* fb0, __bf16* fb1, __bf16* fbT0, __bf16* fbT1,
                      float* sstack, float* logits,
                      float* evo_ru, float* evo_c, float* ubuf) {
  concat_in_kernel<<<(BB * NN + 255) / 256, 256, 0, st>>>(x, xbs, state, fb0, fbT0);
  run_evo(st, Sbf, fb0, fb1, fbT0, fbT1, 2 * HD, pru, sstack, logits, evo_ru);
  size_t nh = (size_t)BB * NN * HD;
  gru_ru_kernel<<<(unsigned)((nh + 255) / 256), 256, 0, st>>>(evo_ru, state, x, xbs,
                                                              fb0, fbT0, ubuf);
  run_evo(st, Sbf, fb0, fb1, fbT0, fbT1, HD, pcand, sstack, logits, evo_c);
  gru_update_kernel<<<(unsigned)((nh + 255) / 256), 256, 0, st>>>(evo_c, ubuf, state);
}

extern "C" void kernel_launch(void* const* d_in, const int* in_sizes, int n_in,
                              void* d_out, int out_size, void* d_ws, size_t ws_size,
                              hipStream_t stream) {
  (void)n_in; (void)out_size; (void)ws_size;
  auto F = [&](int i) { return (const float*)d_in[i]; };

  // ---- input mapping: dict-insertion order vs JAX sorted-pytree order ----
  const float *inputs, *nv1, *nv2, *w1, *w2, *b1, *b2, *outW, *outb;
  EvoP enc_ru, enc_cand, dec_ru, dec_cand;
  auto evo_sorted = [&](int base, EvoP& p) {   // W0..2, attW, attb, b0..2
    for (int i = 0; i < 3; ++i) p.W[i] = F(base + i);
    p.attW = F(base + 3); p.attb = F(base + 4);
    for (int i = 0; i < 3; ++i) p.bvec[i] = F(base + 5 + i);
  };
  auto evo_insert = [&](int base, EvoP& p) {   // W0..2, b0..2, attW, attb
    for (int i = 0; i < 3; ++i) p.W[i] = F(base + i);
    for (int i = 0; i < 3; ++i) p.bvec[i] = F(base + 3 + i);
    p.attW = F(base + 6); p.attb = F(base + 7);
  };
  if (in_sizes[1] == 64) {  // sorted pytree order
    inputs = F(0); b1 = F(1); b2 = F(2);
    evo_sorted(3, dec_cand); evo_sorted(11, dec_ru);
    outW = F(19); outb = F(20);
    evo_sorted(21, enc_cand); evo_sorted(29, enc_ru);
    nv1 = F(37); nv2 = F(38); w1 = F(39); w2 = F(40);
  } else {                  // dict insertion order
    inputs = F(0); nv1 = F(1); nv2 = F(2); w1 = F(3); w2 = F(4); b1 = F(5); b2 = F(6);
    evo_insert(7, enc_ru); evo_insert(15, enc_cand);
    evo_insert(23, dec_ru); evo_insert(31, dec_cand);
    outW = F(39); outb = F(40);
  }

  // ---- workspace carve-up (256B aligned) ----
  char* ws = (char*)d_ws;
  size_t off = 0;
  auto carve = [&](size_t bytes) { void* p = ws + off; off += (bytes + 255) & ~(size_t)255; return p; };
  __bf16* Sbf0 = (__bf16*)carve((size_t)NN * NN * 2);
  __bf16* Sbf1 = (__bf16*)carve((size_t)NN * NN * 2);
  __bf16* Sbf2 = (__bf16*)carve((size_t)NN * NN * 2);
  float* nv1s1 = (float*)carve((size_t)NN * ND * 4);
  float* nv2s1 = (float*)carve((size_t)NN * ND * 4);
  float* nv1s2 = (float*)carve((size_t)NN * ND * 4);
  float* nv2s2 = (float*)carve((size_t)NN * ND * 4);
  __bf16* fb0  = (__bf16*)carve((size_t)BB * NN * 4 * FBLD * 2);
  __bf16* fb1  = (__bf16*)carve((size_t)BB * NN * 4 * FBLD * 2);
  __bf16* fbT0 = (__bf16*)carve((size_t)BB * TPL * FBLD * NN * 2);
  __bf16* fbT1 = (__bf16*)carve((size_t)BB * TPL * FBLD * NN * 2);
  float* sstack = (float*)carve((size_t)BB * 3 * NN * 128 * 4);
  float* evo_ru = (float*)carve((size_t)BB * NN * 128 * 4);
  float* evo_c  = (float*)carve((size_t)BB * NN * HD * 4);
  float* ubuf   = (float*)carve((size_t)BB * NN * HD * 4);
  float* state  = (float*)carve((size_t)BB * NN * HD * 4);
  float* dec_in = (float*)carve((size_t)BB * NN * 4);
  float* logits = (float*)carve((size_t)BB * 3 * 4);

  // per-evo packed weights (bf16, transposed + zero padded)
  EvoP* evos[4] = {&enc_ru, &enc_cand, &dec_ru, &dec_cand};
  int   ods[4]  = {2 * HD, HD, 2 * HD, HD};
  for (int e = 0; e < 4; ++e) {
    EvoP& p = *evos[e];
    int OD = ods[e];
    for (int i = 0; i < 3; ++i) {
      p.d[i]    = (i == 0) ? D0 : OD;
      p.dpad[i] = (i == 0) ? D0PAD : OD;
      p.Wt[i]   = (__bf16*)carve((size_t)OD * 4 * p.dpad[i] * 2);
    }
  }
  for (int e = 0; e < 4; ++e) {
    EvoP& p = *evos[e];
    int OD = ods[e];
    for (int i = 0; i < 3; ++i) {
      int total = OD * 4 * p.dpad[i];
      wt_pack_kernel<<<(total + 255) / 256, 256, 0, stream>>>(p.W[i], p.Wt[i],
                                                              p.d[i], p.dpad[i], OD);
    }
  }

  // ---- supports: softmax(leaky_relu(nv1_s @ nv2_s)) -> bf16 (computed once) ----
  support_kernel<<<NN, 256, 0, stream>>>(nv1, nv2, Sbf0);
  nv_stage_kernel<<<(2 * NN * ND + 255) / 256, 256, 0, stream>>>(nv1, nv2, w1, b1, nv1s1, nv2s1);
  support_kernel<<<NN, 256, 0, stream>>>(nv1s1, nv2s1, Sbf1);
  nv_stage_kernel<<<(2 * NN * ND + 255) / 256, 256, 0, stream>>>(nv1s1, nv2s1, w2, b2, nv1s2, nv2s2);
  support_kernel<<<NN, 256, 0, stream>>>(nv1s2, nv2s2, Sbf2);
  const __bf16* Sbf[3] = {Sbf0, Sbf1, Sbf2};

  // ---- init recurrent state / decoder input ----
  hipMemsetAsync(state, 0, (size_t)BB * NN * HD * 4, stream);
  hipMemsetAsync(dec_in, 0, (size_t)BB * NN * 4, stream);

  // ---- encoder ----
  for (int t = 0; t < TT; ++t) {
    const float* xt = inputs + (size_t)t * NN;     // batch stride T*N
    run_dcgru(stream, Sbf, xt, (size_t)TT * NN, state, enc_ru, enc_cand,
              fb0, fb1, fbT0, fbT1, sstack, logits, evo_ru, evo_c, ubuf);
  }

  // ---- decoder ----
  float* dout = (float*)d_out;
  for (int t = 0; t < NPRED; ++t) {
    run_dcgru(stream, Sbf, dec_in, (size_t)NN, state, dec_ru, dec_cand,
              fb0, fb1, fbT0, fbT1, sstack, logits, evo_ru, evo_c, ubuf);
    dec_out_kernel<<<(BB * NN + 255) / 256, 256, 0, stream>>>(state, outW, outb, dout, dec_in, t);
  }
}
